// BertGAT_47991964565772
// MI455X (gfx1250) — compile-verified, matrix operand order checked by reading, not run on hardware
//
#include <hip/hip_runtime.h>
#include <cstdint>
#include <cstddef>

// ---------------- problem dims (from reference) ----------------
#define V_   50265
#define L_   16
#define D_   768
#define NH_  12
#define FF_  3072
#define NL_  2
#define N_   2048
#define E_   65536
#define NET_ 6
#define HC_  64
#define GH_  4
#define HD_  64
#define NLR  (N_ * L_)          // 32768 rows

typedef unsigned short ushort;

// ---------------- WMMA types ----------------
typedef __bf16 bf16_t;
typedef bf16_t bf16x16 __attribute__((ext_vector_type(16)));
typedef float  f32x8   __attribute__((ext_vector_type(8)));

union FragAB { bf16x16 v; uint4 q[2]; };

__device__ __forceinline__ ushort f2bf(float f) {
  unsigned u = __float_as_uint(f);
  unsigned r = u + 0x7FFFu + ((u >> 16) & 1u);   // round-to-nearest-even
  return (ushort)(r >> 16);
}
__device__ __forceinline__ float gelu_tanh(float x) {
  const float c = 0.7978845608028654f;
  float t = c * (x + 0.044715f * x * x * x);
  return 0.5f * x * (1.0f + tanhf(t));
}

// -------- CDNA5 async copy: 16B global -> LDS per lane (ASYNCcnt tracked) ----
__device__ __forceinline__ void async_copy16(unsigned lds_off, const void* gptr) {
  unsigned long long ga = (unsigned long long)(uintptr_t)gptr;
  asm volatile("global_load_async_to_lds_b128 %0, %1, off"
               :: "v"(lds_off), "v"(ga) : "memory");
}
__device__ __forceinline__ void wait_async0() {
  asm volatile("s_wait_asynccnt 0x0" ::: "memory");
}

// ---------------- WMMA GEMM --------------------------------------------------
// A: [M][K] bf16 row-major. Bt: [N][K] bf16 (pre-transposed weights).
// C = act(A @ B + bias); writes fp32 (Cf) and/or bf16 (Cb).
// Requires M%128==0, N%128==0, K%32==0 (true at all call sites).
#define BM 128
#define BN 128
#define BK 32
#define APITCH 40   // 32 bf16 + 8 pad (80B row pitch, keeps 16B alignment)

__global__ __launch_bounds__(256)
void gemm_bf16_wmma(const ushort* __restrict__ A, const ushort* __restrict__ Bt,
                    const float* __restrict__ bias, float* __restrict__ Cf,
                    ushort* __restrict__ Cb, int M, int N, int K, int act)
{
  __shared__ __align__(16) ushort As[2][BM * APITCH];
  __shared__ __align__(16) ushort Bs[2][BN * APITCH];

  const int tid  = threadIdx.x;
  const int lane = tid & 31;
  const int wave = tid >> 5;
  const int bm = blockIdx.y * BM;
  const int bn = blockIdx.x * BN;

  const int warpRow = (wave & 3) * 32;   // 4 wave-rows of 32
  const int warpCol = (wave >> 2) * 64;  // 2 wave-cols of 64

  const int half = lane >> 4;            // K-half select per WMMA layout
  const int lm   = lane & 15;

  f32x8 acc[2][4];
  #pragma unroll
  for (int i = 0; i < 2; ++i)
    #pragma unroll
    for (int j = 0; j < 4; ++j)
      #pragma unroll
      for (int g = 0; g < 8; ++g) acc[i][j][g] = 0.0f;

  // branch-free staging: each thread copies 2 A chunks + 2 B chunks (16B each)
  auto stage = [&](int kt, int buf) {
    #pragma unroll
    for (int i = 0; i < 2; ++i) {
      int idx = tid + i * 256;                  // 0..511
      int row = idx >> 2, chunk = idx & 3;
      unsigned lds = (unsigned)(uintptr_t)&As[buf][row * APITCH + chunk * 8];
      async_copy16(lds, A + (size_t)(bm + row) * K + kt + chunk * 8);
    }
    #pragma unroll
    for (int i = 0; i < 2; ++i) {
      int idx = tid + i * 256;
      int row = idx >> 2, chunk = idx & 3;
      unsigned lds = (unsigned)(uintptr_t)&Bs[buf][row * APITCH + chunk * 8];
      async_copy16(lds, Bt + (size_t)(bn + row) * K + kt + chunk * 8);
    }
  };

  stage(0, 0);
  wait_async0();
  __syncthreads();

  const int nk = K / BK;
  for (int kt = 0; kt < nk; ++kt) {
    const int buf = kt & 1;
    if (kt + 1 < nk) stage((kt + 1) * BK, buf ^ 1);
    if (kt + 4 < nk) {   // warm GL2 ahead of the async engine
      int kp = (kt + 4) * BK;
      int row = tid >> 1, chunk = tid & 1;     // 128 rows x 2 halves
      __builtin_prefetch(A + (size_t)(bm + row) * K + kp + chunk * 16, 0, 0);
      __builtin_prefetch(Bt + (size_t)(bn + row) * K + kp + chunk * 16, 0, 0);
    }

    FragAB a[2], b[4];
    #pragma unroll
    for (int fr = 0; fr < 2; ++fr) {
      const ushort* p = &As[buf][(warpRow + fr * 16 + lm) * APITCH];
      a[fr].q[0] = *(const uint4*)(p + 8 * half);       // K 0..7  (+8*half)
      a[fr].q[1] = *(const uint4*)(p + 16 + 8 * half);  // K 16..23 (+8*half)
    }
    #pragma unroll
    for (int fc = 0; fc < 4; ++fc) {
      const ushort* p = &Bs[buf][(warpCol + fc * 16 + lm) * APITCH];
      b[fc].q[0] = *(const uint4*)(p + 8 * half);
      b[fc].q[1] = *(const uint4*)(p + 16 + 8 * half);
    }
    #pragma unroll
    for (int fr = 0; fr < 2; ++fr)
      #pragma unroll
      for (int fc = 0; fc < 4; ++fc)
        acc[fr][fc] = __builtin_amdgcn_wmma_f32_16x16x32_bf16(
            false, a[fr].v, false, b[fc].v, (short)0, acc[fr][fc], false, false);

    wait_async0();      // this wave's async copies into buf^1 have landed
    __syncthreads();    // all waves done reading buf / writing buf^1
  }

  // D-tile layout: VGPR g -> M = g + 8*(lane/16), N = lane%16
  #pragma unroll
  for (int fr = 0; fr < 2; ++fr) {
    #pragma unroll
    for (int fc = 0; fc < 4; ++fc) {
      #pragma unroll
      for (int g = 0; g < 8; ++g) {
        int row = bm + warpRow + fr * 16 + g + 8 * half;
        int col = bn + warpCol + fc * 16 + lm;
        float v = acc[fr][fc][g];
        if (bias) v += bias[col];
        if (act == 1) v = gelu_tanh(v);
        else if (act == 2) v = fmaxf(v, 0.0f);
        size_t off = (size_t)row * N + col;
        if (Cf) Cf[off] = v;
        if (Cb) Cb[off] = f2bf(v);
      }
    }
  }
}

// ---------------- weight convert + transpose: W[K][N] f32 -> Wt[N][K] bf16 ---
__global__ void wconv_kernel(const float* __restrict__ W, ushort* __restrict__ Wt,
                             int K, int N) {
  size_t idx = (size_t)blockIdx.x * blockDim.x + threadIdx.x;
  if (idx >= (size_t)K * N) return;
  int k = (int)(idx / N), n = (int)(idx % N);
  Wt[(size_t)n * K + k] = f2bf(W[idx]);
}

// ---------------- embedding --------------------------------------------------
__global__ void embed_kernel(const int* __restrict__ x, const float* __restrict__ tok,
                             const float* __restrict__ pos, float* __restrict__ out)
{
  size_t idx = (size_t)blockIdx.x * blockDim.x + threadIdx.x;
  if (idx >= (size_t)NLR * D_) return;
  int row = (int)(idx / D_);
  int c   = (int)(idx % D_);
  out[idx] = tok[(size_t)x[row] * D_ + c] + pos[(size_t)(row % L_) * D_ + c];
}

// ---------------- layernorm (optional residual), dual fp32 + bf16 output ----
__global__ __launch_bounds__(256)
void ln_kernel(const float* __restrict__ x, const float* __restrict__ r,
               const float* __restrict__ g, const float* __restrict__ b,
               float* __restrict__ out, ushort* __restrict__ outb)
{
  const int row = blockIdx.x;
  const int tid = threadIdx.x;
  __shared__ float sh[256];

  float vals[3];
  float s = 0.0f;
  #pragma unroll
  for (int i = 0; i < 3; ++i) {
    int c = tid + i * 256;
    float v = x[(size_t)row * D_ + c];
    if (r) v += r[(size_t)row * D_ + c];
    vals[i] = v; s += v;
  }
  sh[tid] = s; __syncthreads();
  for (int st = 128; st > 0; st >>= 1) { if (tid < st) sh[tid] += sh[tid + st]; __syncthreads(); }
  float mean = sh[0] * (1.0f / D_);
  __syncthreads();

  float s2 = 0.0f;
  #pragma unroll
  for (int i = 0; i < 3; ++i) { float d = vals[i] - mean; s2 += d * d; }
  sh[tid] = s2; __syncthreads();
  for (int st = 128; st > 0; st >>= 1) { if (tid < st) sh[tid] += sh[tid + st]; __syncthreads(); }
  float rstd = rsqrtf(sh[0] * (1.0f / D_) + 1e-5f);
  __syncthreads();

  #pragma unroll
  for (int i = 0; i < 3; ++i) {
    int c = tid + i * 256;
    float v = (vals[i] - mean) * rstd * g[c] + b[c];
    out[(size_t)row * D_ + c] = v;
    outb[(size_t)row * D_ + c] = f2bf(v);
  }
}

// ---------------- attention (L=16, HD=64), bf16 output -----------------------
__global__ __launch_bounds__(256)
void attn_kernel(const float* __restrict__ q, const float* __restrict__ k,
                 const float* __restrict__ v, ushort* __restrict__ o)
{
  const int n = blockIdx.x;
  const int tid = threadIdx.x;
  __shared__ float s[NH_][L_][L_];

  #pragma unroll 1
  for (int i = 0; i < 12; ++i) {
    int e  = tid * 12 + i;             // 3072 scores
    int h  = e >> 8;
    int qi = (e >> 4) & 15;
    int kj = e & 15;
    const float* qp = q + ((size_t)(n * L_ + qi) * D_) + h * HD_;
    const float* kp = k + ((size_t)(n * L_ + kj) * D_) + h * HD_;
    float acc = 0.0f;
    for (int d = 0; d < HD_; ++d) acc += qp[d] * kp[d];
    s[h][qi][kj] = acc * 0.125f;       // 1/sqrt(64)
  }
  __syncthreads();

  if (tid < NH_ * L_) {
    int h = tid >> 4, qi = tid & 15;
    float m = -1e30f;
    for (int j = 0; j < L_; ++j) m = fmaxf(m, s[h][qi][j]);
    float den = 0.0f;
    for (int j = 0; j < L_; ++j) { float e = expf(s[h][qi][j] - m); s[h][qi][j] = e; den += e; }
    float inv = 1.0f / den;
    for (int j = 0; j < L_; ++j) s[h][qi][j] *= inv;
  }
  __syncthreads();

  #pragma unroll 1
  for (int i = 0; i < 48; ++i) {
    int e   = tid * 48 + i;            // 12288 outputs
    int l   = e / D_;
    int col = e % D_;
    int h   = col >> 6;
    float acc = 0.0f;
    for (int j = 0; j < L_; ++j)
      acc += s[h][l][j] * v[(size_t)(n * L_ + j) * D_ + col];
    o[(size_t)(n * L_ + l) * D_ + col] = f2bf(acc);
  }
}

// ---------------- small utility kernels -------------------------------------
__global__ void fill_f32(float* p, float v, size_t n) {
  size_t i = (size_t)blockIdx.x * blockDim.x + threadIdx.x;
  if (i < n) p[i] = v;
}
__global__ void fill_i32(int* p, int v, size_t n) {
  size_t i = (size_t)blockIdx.x * blockDim.x + threadIdx.x;
  if (i < n) p[i] = v;
}
__global__ void cls_kernel(const ushort* __restrict__ hb, ushort* __restrict__ feats) {
  size_t i = (size_t)blockIdx.x * blockDim.x + threadIdx.x;
  if (i >= (size_t)N_ * D_) return;
  int n = (int)(i / D_), c = (int)(i % D_);
  feats[i] = hb[(size_t)n * L_ * D_ + c];
}

// ---------------- GAT support ------------------------------------------------
__device__ __forceinline__ int f2ord(float f) {
  int i = __float_as_int(f);
  return (i >= 0) ? i : (i ^ 0x7FFFFFFF);
}
__device__ __forceinline__ float ord2f(int s) {
  int i = (s >= 0) ? s : (s ^ 0x7FFFFFFF);
  return __int_as_float(i);
}

__global__ void deg_kernel(const int* __restrict__ dst, float* __restrict__ deg) {
  int e = blockIdx.x * blockDim.x + threadIdx.x;
  if (e < E_) atomicAdd(&deg[dst[e]], 1.0f);
}

__global__ void wvec_kernel(const float* __restrict__ We, const float* __restrict__ ae,
                            float* __restrict__ wvec, int H, int C) {
  int idx = blockIdx.x * blockDim.x + threadIdx.x;
  if (idx >= D_ * H) return;
  int d = idx / H, h = idx % H, F = H * C;
  float s = 0.0f;
  for (int c = 0; c < C; ++c) s += We[(size_t)d * F + h * C + c] * ae[h * C + c];
  wvec[d * H + h] = s;
}

__global__ void aetype_kernel(const float* __restrict__ ee, const float* __restrict__ wvec,
                              float* __restrict__ aetype, int H) {
  int idx = blockIdx.x * blockDim.x + threadIdx.x;
  if (idx >= NET_ * H) return;
  int t = idx / H, h = idx % H;
  float s = 0.0f;
  for (int d = 0; d < D_; ++d) s += ee[(size_t)t * D_ + d] * wvec[d * H + h];
  aetype[idx] = s;
}

__global__ void sumae_kernel(const int* __restrict__ dst, const int* __restrict__ et,
                             const float* __restrict__ aetype, float* __restrict__ sumae, int H) {
  int e = blockIdx.x * blockDim.x + threadIdx.x;
  if (e >= E_) return;
  int t = et[e], dd = dst[e];
  for (int h = 0; h < H; ++h) atomicAdd(&sumae[dd * H + h], aetype[t * H + h]);
}

__global__ void headscore_kernel(const float* __restrict__ xh, const float* __restrict__ as,
                                 const float* __restrict__ ad, float* __restrict__ osrc,
                                 float* __restrict__ odst, int H, int C) {
  int idx = blockIdx.x * blockDim.x + threadIdx.x;
  if (idx >= N_ * H) return;
  int n = idx / H, h = idx % H, F = H * C;
  float s1 = 0.0f, s2 = 0.0f;
  for (int c = 0; c < C; ++c) {
    float v = xh[(size_t)n * F + h * C + c];
    s1 += v * as[h * C + c];
    s2 += v * ad[h * C + c];
  }
  osrc[idx] = s1; odst[idx] = s2;
}

__global__ void alpha_kernel(const int* __restrict__ src, const int* __restrict__ dst,
                             const int* __restrict__ et, const float* __restrict__ asrc,
                             const float* __restrict__ adst, const float* __restrict__ aetype,
                             const float* __restrict__ sumae, const float* __restrict__ deg,
                             float* __restrict__ alpha, int* __restrict__ mbuf, int H) {
  int e = blockIdx.x * blockDim.x + threadIdx.x;
  if (e >= E_ + N_) return;
  int s, dd; float ae[GH_];
  if (e < E_) {
    s = src[e]; dd = dst[e]; int t = et[e];
    for (int h = 0; h < H; ++h) ae[h] = aetype[t * H + h];
  } else {
    int n = e - E_; s = n; dd = n;
    float dg = fmaxf(deg[n], 1.0f);
    for (int h = 0; h < H; ++h) ae[h] = sumae[n * H + h] / dg;
  }
  for (int h = 0; h < H; ++h) {
    float a = asrc[s * H + h] + adst[dd * H + h] + ae[h];
    a = (a > 0.0f) ? a : 0.2f * a;           // leaky_relu(0.2)
    alpha[(size_t)e * H + h] = a;
    atomicMax(&mbuf[dd * H + h], f2ord(a));
  }
}

__global__ void expden_kernel(const int* __restrict__ dst, float* __restrict__ alpha,
                              const int* __restrict__ mbuf, float* __restrict__ den, int H) {
  int e = blockIdx.x * blockDim.x + threadIdx.x;
  if (e >= E_ + N_) return;
  int dd = (e < E_) ? dst[e] : (e - E_);
  for (int h = 0; h < H; ++h) {
    float ex = expf(alpha[(size_t)e * H + h] - ord2f(mbuf[dd * H + h]));
    alpha[(size_t)e * H + h] = ex;
    atomicAdd(&den[dd * H + h], ex);
  }
}

__global__ void scatter_kernel(const int* __restrict__ src, const int* __restrict__ dst,
                               const float* __restrict__ xh, const float* __restrict__ ex,
                               const float* __restrict__ den, float* __restrict__ out,
                               int H, int C) {
  int e = blockIdx.x * blockDim.x + threadIdx.x;
  if (e >= E_ + N_) return;
  int s = (e < E_) ? src[e] : (e - E_);
  int dd = (e < E_) ? dst[e] : (e - E_);
  int F = H * C;
  for (int h = 0; h < H; ++h) {
    float w = ex[(size_t)e * H + h] / den[dd * H + h];
    for (int c = 0; c < C; ++c)
      atomicAdd(&out[(size_t)dd * F + h * C + c], xh[(size_t)s * F + h * C + c] * w);
  }
}

__global__ void biasact_kernel(const float* __restrict__ in, const float* __restrict__ bias,
                               float* __restrict__ out, ushort* __restrict__ outb,
                               int n, int F, int relu) {
  size_t i = (size_t)blockIdx.x * blockDim.x + threadIdx.x;
  if (i >= (size_t)n * F) return;
  float v = in[i] + bias[i % F];
  if (relu) v = fmaxf(v, 0.0f);
  out[i] = v;
  if (outb) outb[i] = f2bf(v);
}

__global__ void matvec_kernel(const float* __restrict__ x, const float* __restrict__ W,
                              float* __restrict__ out, int Kdim) {
  int n = blockIdx.x * blockDim.x + threadIdx.x;
  if (n >= N_) return;
  float s = 0.0f;
  for (int k = 0; k < Kdim; ++k) s += x[(size_t)n * Kdim + k] * W[k];
  out[n] = s;
}

// ---------------- host orchestration ----------------------------------------
static inline int cdiv(long long a, long long b) { return (int)((a + b - 1) / b); }

extern "C" void kernel_launch(void* const* d_in, const int* in_sizes, int n_in,
                              void* d_out, int out_size, void* d_ws, size_t ws_size,
                              hipStream_t stream) {
  (void)in_sizes; (void)n_in; (void)out_size; (void)ws_size;

  const int*   x      = (const int*)d_in[0];
  const int*   eidx   = (const int*)d_in[1];
  const int*   eattr  = (const int*)d_in[2];
  const float* tok    = (const float*)d_in[3];
  const float* pos    = (const float*)d_in[4];
  const float* emb_g  = (const float*)d_in[5];
  const float* emb_b  = (const float*)d_in[6];
  const float* eemb   = (const float*)d_in[39];
  float* out = (float*)d_out;

  const int* esrc = eidx;
  const int* edst = eidx + E_;

  // ---- workspace layout (byte allocator, 256B aligned) ----
  char* base = (char*)d_ws;
  size_t o = 0;
  auto allocB = [&](size_t bytes) { char* p = base + o; o += (bytes + 255) & ~(size_t)255; return p; };

  float*  h     = (float*)allocB((size_t)NLR * D_ * 4);
  float*  big   = (float*)allocB((size_t)NLR * FF_ * 4);  // q|k|v fp32; fbig_bf aliases
  char*   t0r   = allocB((size_t)NLR * D_ * 4);           // o_bf region
  float*  t1    = (float*)allocB((size_t)NLR * D_ * 4);
  ushort* h_bf  = (ushort*)allocB((size_t)NLR * D_ * 2);
  ushort* wbf   = (ushort*)allocB((size_t)16 * 1024 * 1024 * 2);  // bf16 weights
  ushort* feats = (ushort*)allocB((size_t)N_ * D_ * 2);
  float*  xh    = (float*)allocB((size_t)N_ * 256 * 4);
  float*  x2    = (float*)allocB((size_t)N_ * 256 * 4);
  ushort* x2b   = (ushort*)allocB((size_t)N_ * 256 * 2);
  float*  gout  = (float*)allocB((size_t)N_ * 256 * 4);
  float*  asrc  = (float*)allocB((size_t)N_ * GH_ * 4);
  float*  adst  = (float*)allocB((size_t)N_ * GH_ * 4);
  float*  wvec  = (float*)allocB((size_t)D_ * GH_ * 4);
  float*  aetype= (float*)allocB((size_t)NET_ * GH_ * 4);
  float*  deg   = (float*)allocB((size_t)N_ * 4);
  float*  sumae = (float*)allocB((size_t)N_ * GH_ * 4);
  float*  den   = (float*)allocB((size_t)N_ * GH_ * 4);
  int*    mbuf  = (int*)allocB((size_t)N_ * GH_ * 4);
  float*  alpha = (float*)allocB((size_t)(E_ + N_) * GH_ * 4);

  float*  qb = big;
  float*  kb = big + (size_t)NLR * D_;
  float*  vb = big + 2 * (size_t)NLR * D_;
  ushort* fbig_bf = (ushort*)big;          // FFN hidden bf16 (after qkv dead)
  ushort* o_bf = (ushort*)t0r;

  auto F = [&](int i) { return (const float*)d_in[i]; };

  // ---- convert + transpose all GEMM weights to bf16 [N][K] ----
  size_t woff = 0;
  auto wconv = [&](int i, int K, int N) {
    ushort* p = wbf + woff; woff += (size_t)K * N;
    wconv_kernel<<<cdiv((size_t)K * N, 256), 256, 0, stream>>>(F(i), p, K, N);
    return p;
  };
  ushort *Wq[NL_], *Wk[NL_], *Wv[NL_], *Wo[NL_], *W1[NL_], *W2[NL_];
  for (int li = 0; li < NL_; ++li) {
    int B = 7 + 16 * li;
    Wq[li] = wconv(B + 0, D_, D_);
    Wk[li] = wconv(B + 2, D_, D_);
    Wv[li] = wconv(B + 4, D_, D_);
    Wo[li] = wconv(B + 6, D_, D_);
    W1[li] = wconv(B + 10, D_, FF_);
    W2[li] = wconv(B + 12, FF_, D_);
  }
  ushort* Wg0 = wconv(40, D_, 256);
  ushort* Wg1 = wconv(46, 256, 256);

  auto gemm = [&](const ushort* A, const ushort* Bt, const float* bias,
                  float* Cf, ushort* Cb, int M, int Nn, int K, int act) {
    dim3 g(Nn / BN, M / BM);
    gemm_bf16_wmma<<<g, 256, 0, stream>>>(A, Bt, bias, Cf, Cb, M, Nn, K, act);
  };

  // ---- BERT ----
  embed_kernel<<<cdiv((size_t)NLR * D_, 256), 256, 0, stream>>>(x, tok, pos, t1);
  ln_kernel<<<NLR, 256, 0, stream>>>(t1, nullptr, emb_g, emb_b, h, h_bf);

  for (int li = 0; li < NL_; ++li) {
    int B = 7 + 16 * li;
    gemm(h_bf, Wq[li], F(B + 1), qb, nullptr, NLR, D_, D_, 0);
    gemm(h_bf, Wk[li], F(B + 3), kb, nullptr, NLR, D_, D_, 0);
    gemm(h_bf, Wv[li], F(B + 5), vb, nullptr, NLR, D_, D_, 0);
    attn_kernel<<<N_, 256, 0, stream>>>(qb, kb, vb, o_bf);
    gemm(o_bf, Wo[li], F(B + 7), t1, nullptr, NLR, D_, D_, 0);
    ln_kernel<<<NLR, 256, 0, stream>>>(h, t1, F(B + 8), F(B + 9), h, h_bf);
    gemm(h_bf, W1[li], F(B + 11), nullptr, fbig_bf, NLR, FF_, D_, 1);   // gelu
    gemm(fbig_bf, W2[li], F(B + 13), t1, nullptr, NLR, D_, FF_, 0);
    ln_kernel<<<NLR, 256, 0, stream>>>(h, t1, F(B + 14), F(B + 15), h, h_bf);
  }

  // ---- GAT ----
  cls_kernel<<<cdiv((size_t)N_ * D_, 256), 256, 0, stream>>>(h_bf, feats);
  fill_f32<<<cdiv(N_, 256), 256, 0, stream>>>(deg, 0.0f, N_);
  deg_kernel<<<cdiv(E_, 256), 256, 0, stream>>>(edst, deg);

  const int ORD_NEG_INF = (int)(0xFF800000u ^ 0x7FFFFFFFu);  // f2ord(-inf)

  const ushort* xin_bf = feats;
  int Fin = D_;
  for (int li = 0; li < 3; ++li) {
    int gb = 40 + 6 * li;
    int H = (li < 2) ? GH_ : 1;
    int C = (li < 2) ? HC_ : 1;
    int Fo = H * C;

    if (li == 0)      gemm(xin_bf, Wg0, nullptr, xh, nullptr, N_, 256, Fin, 0);
    else if (li == 1) gemm(xin_bf, Wg1, nullptr, xh, nullptr, N_, 256, Fin, 0);
    else matvec_kernel<<<cdiv(N_, 256), 256, 0, stream>>>(x2, F(gb + 0), xh, Fin);

    wvec_kernel<<<cdiv(D_ * H, 256), 256, 0, stream>>>(F(gb + 1), F(gb + 4), wvec, H, C);
    aetype_kernel<<<1, 64, 0, stream>>>(eemb, wvec, aetype, H);
    headscore_kernel<<<cdiv(N_ * H, 256), 256, 0, stream>>>(xh, F(gb + 2), F(gb + 3),
                                                            asrc, adst, H, C);
    fill_f32<<<cdiv(N_ * H, 256), 256, 0, stream>>>(sumae, 0.0f, (size_t)N_ * H);
    sumae_kernel<<<cdiv(E_, 256), 256, 0, stream>>>(edst, eattr, aetype, sumae, H);

    fill_i32<<<cdiv(N_ * H, 256), 256, 0, stream>>>(mbuf, ORD_NEG_INF, (size_t)N_ * H);
    alpha_kernel<<<cdiv(E_ + N_, 256), 256, 0, stream>>>(esrc, edst, eattr, asrc, adst,
                                                         aetype, sumae, deg, alpha, mbuf, H);
    fill_f32<<<cdiv(N_ * H, 256), 256, 0, stream>>>(den, 0.0f, (size_t)N_ * H);
    expden_kernel<<<cdiv(E_ + N_, 256), 256, 0, stream>>>(edst, alpha, mbuf, den, H);
    fill_f32<<<cdiv((size_t)N_ * Fo, 256), 256, 0, stream>>>(gout, 0.0f, (size_t)N_ * Fo);
    scatter_kernel<<<cdiv(E_ + N_, 256), 256, 0, stream>>>(esrc, edst, xh, alpha, den,
                                                           gout, H, C);
    if (li < 2) {
      biasact_kernel<<<cdiv((size_t)N_ * Fo, 256), 256, 0, stream>>>(gout, F(gb + 5), x2,
                                                                     x2b, N_, Fo, 1);
      xin_bf = x2b; Fin = Fo;
    } else {
      biasact_kernel<<<cdiv(N_, 256), 256, 0, stream>>>(gout, F(gb + 5), out, nullptr,
                                                        N_, 1, 0);
    }
  }
}
